// Seq2SeqModel_12128987644240
// MI455X (gfx1250) — compile-verified, matrix operand order checked by reading, not run on hardware
//
#include <hip/hip_runtime.h>
#include <hip/hip_bf16.h>

typedef __attribute__((ext_vector_type(16))) _Float16 v16h;
typedef __attribute__((ext_vector_type(8)))  _Float16 v8h;
typedef __attribute__((ext_vector_type(8)))  float    v8f;

#define VOCAB 32000
#define EMBED 256
#define HID   512
#define BATCH 32
#define TE    128
#define TD    128
#define G4H   2048   // 4*HID

union Frag { v16h v; v8h h[2]; };

__device__ __forceinline__ float sigf(float x) { return 1.0f / (1.0f + __expf(-x)); }

// ---- convert f32 weight [K,N] -> f16 transposed [N,K] ----
__global__ void k_cvt_t(const float* __restrict__ in, _Float16* __restrict__ out,
                        int K, int N) {
  long gid = (long)blockIdx.x * blockDim.x + threadIdx.x;
  long total = (long)K * N;
  if (gid >= total) return;
  int n = (int)(gid / K), k = (int)(gid % K);
  out[gid] = (_Float16)in[(long)k * N + n];
}

// ---- embedding gather -> f16 rows ----
__global__ void k_gather(const int* __restrict__ ids, const float* __restrict__ emb,
                         _Float16* __restrict__ out, int E) {
  int row = blockIdx.x;
  int tok = ids[row];
  for (int e = threadIdx.x; e < E; e += blockDim.x)
    out[(long)row * E + e] = (_Float16)emb[(long)tok * E + e];
}

// ---- small WMMA GEMM (one 16x16 tile per wave), used only for M=32 case ----
// grid = (N/64, M/32), block = 256
__global__ void k_gemm(const _Float16* __restrict__ A,
                       const _Float16* __restrict__ Bt,
                       const float* __restrict__ bias,
                       float* __restrict__ C,
                       int M, int N, int K) {
  int lane = threadIdx.x & 31;
  int wave = threadIdx.x >> 5;
  int row0 = blockIdx.y * 32 + (wave >> 2) * 16;
  int col0 = blockIdx.x * 64 + (wave & 3) * 16;
  int l15 = lane & 15;
  int hi  = lane >> 4;
  int b0  = hi * 8;
  const _Float16* arow = A  + (long)(row0 + l15) * K;
  const _Float16* brow = Bt + (long)(col0 + l15) * K;
  v8f acc;
  float bv = bias ? bias[col0 + l15] : 0.0f;
#pragma unroll
  for (int i = 0; i < 8; i++) acc[i] = bv;
  for (int kk = 0; kk < K; kk += 32) {
    Frag a, b;
    a.h[0] = *(const v8h*)(arow + kk + b0);
    a.h[1] = *(const v8h*)(arow + kk + 16 + b0);
    b.h[0] = *(const v8h*)(brow + kk + b0);
    b.h[1] = *(const v8h*)(brow + kk + 16 + b0);
    acc = __builtin_amdgcn_wmma_f32_16x16x32_f16(false, a.v, false, b.v,
                                                 (short)0, acc, false, false);
  }
  float* crow = C + (long)(row0 + hi * 8) * N + col0 + l15;
#pragma unroll
  for (int r = 0; r < 8; r++) crow[(long)r * N] = acc[r];
}

// ---- register-blocked WMMA GEMM: 32x64 per wave (2x4 tiles, 8 accumulators) ----
// block = 256 (8 waves arranged 4M x 2N) -> 128x128 block tile.
// grid = (N/128, M/128). Requires M%128==0, N%128==0, K%32==0.
__global__ void k_gemm_big(const _Float16* __restrict__ A,
                           const _Float16* __restrict__ Bt,
                           const float* __restrict__ bias,
                           float* __restrict__ C,
                           int M, int N, int K) {
  int lane = threadIdx.x & 31;
  int wave = threadIdx.x >> 5;
  int row0 = blockIdx.y * 128 + (wave >> 1) * 32;  // 4 waves along M
  int col0 = blockIdx.x * 128 + (wave & 1) * 64;   // 2 waves along N
  int l15 = lane & 15;
  int hi  = lane >> 4;
  int b0  = hi * 8;
  const _Float16* arow0 = A + (long)(row0 + l15) * K;        // M-tile 0
  const _Float16* arow1 = A + (long)(row0 + 16 + l15) * K;   // M-tile 1
  const _Float16* bbase = Bt + (long)(col0 + l15) * K;       // N-tile stride = 16*K

  v8f acc[2][4];
#pragma unroll
  for (int ni = 0; ni < 4; ni++) {
    float bv = bias ? bias[col0 + ni * 16 + l15] : 0.0f;
#pragma unroll
    for (int mi = 0; mi < 2; mi++)
#pragma unroll
      for (int r = 0; r < 8; r++) acc[mi][ni][r] = bv;
  }

  for (int kk = 0; kk < K; kk += 32) {
    Frag a0, a1, b[4];
    a0.h[0] = *(const v8h*)(arow0 + kk + b0);
    a0.h[1] = *(const v8h*)(arow0 + kk + 16 + b0);
    a1.h[0] = *(const v8h*)(arow1 + kk + b0);
    a1.h[1] = *(const v8h*)(arow1 + kk + 16 + b0);
#pragma unroll
    for (int ni = 0; ni < 4; ni++) {
      const _Float16* bp = bbase + (long)ni * 16 * K + kk + b0;
      b[ni].h[0] = *(const v8h*)(bp);
      b[ni].h[1] = *(const v8h*)(bp + 16);
    }
    if (kk + 64 < K) __builtin_prefetch(bbase + kk + 64, 0, 1);  // global_prefetch_b8
#pragma unroll
    for (int ni = 0; ni < 4; ni++) {
      acc[0][ni] = __builtin_amdgcn_wmma_f32_16x16x32_f16(false, a0.v, false, b[ni].v,
                                                          (short)0, acc[0][ni], false, false);
      acc[1][ni] = __builtin_amdgcn_wmma_f32_16x16x32_f16(false, a1.v, false, b[ni].v,
                                                          (short)0, acc[1][ni], false, false);
    }
  }

#pragma unroll
  for (int mi = 0; mi < 2; mi++) {
    float* crow = C + (long)(row0 + mi * 16 + hi * 8) * N + col0 + l15;
#pragma unroll
    for (int ni = 0; ni < 4; ni++)
#pragma unroll
      for (int r = 0; r < 8; r++) crow[(long)r * N + ni * 16] = acc[mi][ni][r];
  }
}

// ---- full LSTM recurrence, single persistent workgroup (32 waves) ----
__global__ void __launch_bounds__(1024)
k_lstm(const float* __restrict__ xp,      // [B*T, 4H] = x@W + b
       const _Float16* __restrict__ Ut,   // [4H, H]  = U^T (f16)
       _Float16* __restrict__ hs,         // [B*T, H] all h_t (f16)
       const _Float16* __restrict__ h0,   // [B,H] or null (zeros)
       const float* __restrict__ c0,      // [B,H] or null (zeros)
       _Float16* __restrict__ hT,         // [B,H] final h or null
       float* __restrict__ cT,            // [B,H] final c or null
       int T) {
  __shared__ _Float16 hbuf[2][BATCH][HID];   // 64 KB double buffer
  _Float16* hflat = &hbuf[0][0][0];
  int tid = threadIdx.x;
  int lane = tid & 31, wave = tid >> 5;
  int l15 = lane & 15, hi = lane >> 4, b0 = hi * 8;
  int col = wave * 16 + l15;                 // this wave's HID column slice
  for (int i = tid; i < BATCH * HID; i += 1024)
    hflat[i] = h0 ? h0[i] : (_Float16)0.0f;
  float cs[2][8];
#pragma unroll
  for (int p = 0; p < 2; p++)
#pragma unroll
    for (int r = 0; r < 8; r++) {
      int m = p * 16 + hi * 8 + r;
      cs[p][r] = c0 ? c0[m * HID + col] : 0.0f;
    }
  __syncthreads();
  for (int t = 0; t < T; t++) {
    int cur = t & 1, nxt = cur ^ 1;
#pragma unroll
    for (int p = 0; p < 2; p++) {           // two 16-row batch tiles
      v8f acc[4];
#pragma unroll
      for (int g = 0; g < 4; g++)
#pragma unroll
        for (int r = 0; r < 8; r++) {
          int m = p * 16 + hi * 8 + r;
          acc[g][r] = xp[((long)m * T + t) * G4H + g * HID + col];
        }
      for (int kk = 0; kk < HID; kk += 32) {
        Frag a;
        const _Float16* ap = &hbuf[cur][p * 16 + l15][kk + b0];
        a.h[0] = *(const v8h*)(ap);
        a.h[1] = *(const v8h*)(ap + 16);
#pragma unroll
        for (int g = 0; g < 4; g++) {
          Frag b;
          const _Float16* bp = Ut + (long)(g * HID + col) * HID + kk + b0;
          b.h[0] = *(const v8h*)(bp);
          b.h[1] = *(const v8h*)(bp + 16);
          acc[g] = __builtin_amdgcn_wmma_f32_16x16x32_f16(false, a.v, false, b.v,
                                                          (short)0, acc[g], false, false);
        }
      }
#pragma unroll
      for (int r = 0; r < 8; r++) {
        int m = p * 16 + hi * 8 + r;
        float iv = sigf(acc[0][r]);
        float fv = sigf(acc[1][r]);
        float gv = tanhf(acc[2][r]);
        float ov = sigf(acc[3][r]);
        float c = fv * cs[p][r] + iv * gv;
        cs[p][r] = c;
        _Float16 hh = (_Float16)(ov * tanhf(c));
        hbuf[nxt][m][col] = hh;
        hs[((long)m * T + t) * HID + col] = hh;
      }
    }
    __syncthreads();
  }
  if (hT) {
    int fin = T & 1;
    for (int i = tid; i < BATCH * HID; i += 1024)
      hT[i] = hflat[(long)fin * BATCH * HID + i];
  }
  if (cT) {
#pragma unroll
    for (int p = 0; p < 2; p++)
#pragma unroll
      for (int r = 0; r < 8; r++) {
        int m = p * 16 + hi * 8 + r;
        cT[m * HID + col] = cs[p][r];
      }
  }
}

// ---- score[b,t] = V . tanh(hW1[b] + encW2[b,t]) ; 1 wave per (b,t) ----
__global__ void k_score(const float* __restrict__ hW1, const float* __restrict__ encW2,
                        const float* __restrict__ V, float* __restrict__ score) {
  int lane = threadIdx.x & 31, wave = threadIdx.x >> 5;
  int pair = blockIdx.x * 8 + wave;         // b*TE + t
  int b = pair >> 7;
  float s = 0.f;
  for (int j = lane; j < HID; j += 32)
    s += tanhf(hW1[b * HID + j] + encW2[(long)pair * HID + j]) * V[j];
  for (int off = 16; off; off >>= 1) s += __shfl_down(s, off, 32);
  if (lane == 0) score[pair] = s;
}

// ---- softmax over TE + context[b] = sum_t attn*enc_out ; 1 block/batch ----
__global__ void k_softmax_ctx(const float* __restrict__ score,
                              const _Float16* __restrict__ enc_out,
                              _Float16* __restrict__ ctx) {
  __shared__ float sc[TE];
  __shared__ float red[TE];
  int t = threadIdx.x, b = blockIdx.x;
  float s = score[b * TE + t];
  sc[t] = s; red[t] = s; __syncthreads();
  for (int st = TE / 2; st; st >>= 1) {
    if (t < st) red[t] = fmaxf(red[t], red[t + st]);
    __syncthreads();
  }
  float e = __expf(sc[t] - red[0]);
  __syncthreads();
  red[t] = e; __syncthreads();
  for (int st = TE / 2; st; st >>= 1) {
    if (t < st) red[t] += red[t + st];
    __syncthreads();
  }
  float inv = 1.0f / red[0];
  __syncthreads();
  sc[t] = e * inv; __syncthreads();
#pragma unroll
  for (int k = 0; k < HID / TE; k++) {
    int c = t + TE * k;
    float a = 0.f;
    for (int tt = 0; tt < TE; tt++)
      a += sc[tt] * (float)enc_out[((long)b * TE + tt) * HID + c];
    ctx[b * HID + c] = (_Float16)a;
  }
}

// ---- decoder input: [dec_emb | context] f16 [B*TD, EMBED+HID] ----
__global__ void k_dec_in(const int* __restrict__ ids, const float* __restrict__ emb,
                         const _Float16* __restrict__ ctx, _Float16* __restrict__ out) {
  long gid = (long)blockIdx.x * blockDim.x + threadIdx.x;
  long total = (long)BATCH * TD * (EMBED + HID);
  if (gid >= total) return;
  int row  = (int)(gid / (EMBED + HID));
  int colc = (int)(gid % (EMBED + HID));
  int b = row / TD;
  _Float16 v;
  if (colc < EMBED) v = (_Float16)emb[(long)ids[row] * EMBED + colc];
  else              v = ctx[b * HID + (colc - EMBED)];
  out[gid] = v;
}

extern "C" void kernel_launch(void* const* d_in, const int* in_sizes, int n_in,
                              void* d_out, int out_size, void* d_ws, size_t ws_size,
                              hipStream_t stream) {
  const int*   enc_ids = (const int*)d_in[0];
  const int*   dec_ids = (const int*)d_in[1];
  const float* emb   = (const float*)d_in[2];
  const float* W_enc = (const float*)d_in[3];
  const float* U_enc = (const float*)d_in[4];
  const float* b_enc = (const float*)d_in[5];
  const float* W1    = (const float*)d_in[6];
  const float* W2    = (const float*)d_in[7];
  const float* V     = (const float*)d_in[8];
  const float* W_dec = (const float*)d_in[9];
  const float* U_dec = (const float*)d_in[10];
  const float* b_dec = (const float*)d_in[11];
  const float* W_fc  = (const float*)d_in[12];
  const float* b_fc  = (const float*)d_in[13];
  float* out = (float*)d_out;
  (void)in_sizes; (void)n_in; (void)out_size; (void)ws_size;

  char* p = (char*)d_ws;
  auto alloc = [&](size_t bytes) -> void* {
    void* r = (void*)p;
    p += (bytes + 255) & ~(size_t)255;
    return r;
  };
  _Float16* WtE  = (_Float16*)alloc((size_t)G4H * EMBED * 2);
  _Float16* UtE  = (_Float16*)alloc((size_t)G4H * HID * 2);
  _Float16* WtD  = (_Float16*)alloc((size_t)G4H * (EMBED + HID) * 2);
  _Float16* UtD  = (_Float16*)alloc((size_t)G4H * HID * 2);
  _Float16* WtF  = (_Float16*)alloc((size_t)VOCAB * HID * 2);
  _Float16* W1t  = (_Float16*)alloc((size_t)HID * HID * 2);
  _Float16* W2t  = (_Float16*)alloc((size_t)HID * HID * 2);
  _Float16* encA = (_Float16*)alloc((size_t)BATCH * TE * EMBED * 2);
  float*    xpE  = (float*)alloc((size_t)BATCH * TE * G4H * 4);
  _Float16* encO = (_Float16*)alloc((size_t)BATCH * TE * HID * 2);
  _Float16* stH  = (_Float16*)alloc((size_t)BATCH * HID * 2);
  float*    stC  = (float*)alloc((size_t)BATCH * HID * 4);
  float*    hW1  = (float*)alloc((size_t)BATCH * HID * 4);
  float*    eW2  = (float*)alloc((size_t)BATCH * TE * HID * 4);
  float*    sco  = (float*)alloc((size_t)BATCH * TE * 4);
  _Float16* ctx  = (_Float16*)alloc((size_t)BATCH * HID * 2);
  _Float16* decA = (_Float16*)alloc((size_t)BATCH * TD * (EMBED + HID) * 2);
  float*    xpD  = (float*)alloc((size_t)BATCH * TD * G4H * 4);
  _Float16* decO = (_Float16*)alloc((size_t)BATCH * TD * HID * 2);

  auto cvt = [&](const float* src, _Float16* dst, int K, int N) {
    long total = (long)K * N;
    k_cvt_t<<<(int)((total + 255) / 256), 256, 0, stream>>>(src, dst, K, N);
  };
  cvt(W_enc, WtE, EMBED, G4H);
  cvt(U_enc, UtE, HID, G4H);
  cvt(W_dec, WtD, EMBED + HID, G4H);
  cvt(U_dec, UtD, HID, G4H);
  cvt(W_fc,  WtF, HID, VOCAB);
  cvt(W1,    W1t, HID, HID);
  cvt(W2,    W2t, HID, HID);

  k_gather<<<BATCH * TE, 256, 0, stream>>>(enc_ids, emb, encA, EMBED);

  // xp_enc = encA @ W_enc + b_enc  [4096,2048]
  k_gemm_big<<<dim3(G4H / 128, (BATCH * TE) / 128), 256, 0, stream>>>(
      encA, WtE, b_enc, xpE, BATCH * TE, G4H, EMBED);
  // encoder LSTM recurrence (serial chain, 1 workgroup, 32 waves)
  k_lstm<<<1, 1024, 0, stream>>>(xpE, UtE, encO, nullptr, nullptr, stH, stC, TE);
  // Bahdanau attention
  k_gemm<<<dim3(HID / 64, 1), 256, 0, stream>>>(stH, W1t, nullptr, hW1, BATCH, HID, HID);
  k_gemm_big<<<dim3(HID / 128, (BATCH * TE) / 128), 256, 0, stream>>>(
      encO, W2t, nullptr, eW2, BATCH * TE, HID, HID);
  k_score<<<(BATCH * TE) / 8, 256, 0, stream>>>(hW1, eW2, V, sco);
  k_softmax_ctx<<<BATCH, TE, 0, stream>>>(sco, encO, ctx);
  // decoder input concat + xp_dec
  {
    long total = (long)BATCH * TD * (EMBED + HID);
    k_dec_in<<<(int)((total + 255) / 256), 256, 0, stream>>>(dec_ids, emb, ctx, decA);
  }
  k_gemm_big<<<dim3(G4H / 128, (BATCH * TD) / 128), 256, 0, stream>>>(
      decA, WtD, b_dec, xpD, BATCH * TD, G4H, EMBED + HID);
  // decoder LSTM recurrence (init from encoder final state)
  k_lstm<<<1, 1024, 0, stream>>>(xpD, UtD, decO, stH, stC, nullptr, nullptr, TD);
  // logits = dec_out @ W_fc + b_fc  [4096,32000] -> d_out (dominant GEMM)
  k_gemm_big<<<dim3(VOCAB / 128, (BATCH * TD) / 128), 256, 0, stream>>>(
      decO, WtF, b_fc, out, BATCH * TD, VOCAB, HID);
}
